// Temporal_Brain_Layer_61959198212469
// MI455X (gfx1250) — compile-verified, matrix operand order checked by reading, not run on hardware
//
#include <hip/hip_runtime.h>
#include <hip/hip_bf16.h>

typedef _Float16 v16h __attribute__((ext_vector_type(16)));
typedef _Float16 v8h  __attribute__((ext_vector_type(8)));
typedef _Float16 v4h  __attribute__((ext_vector_type(4)));
typedef float    v8f  __attribute__((ext_vector_type(8)));

#define NB 8          // batches per workgroup (1 per wave)
#define THREADS 256   // 8 waves
#define NTILES 64     // 48 qkv weight tiles + 16 out-proj weight tiles
#define TILE_HALF 4096 // 16 rows x 256 cols f16 per tile (8 KB)

// Row bitmasks of the fixed 26x26 anatomical attention pattern.
__constant__ unsigned MASKROW[26] = {
    0x210000Fu, 0x210000Fu, 0x210000Fu, 0x210000Fu,   // room 5 (0-3)
    0x26000F0u, 0x26000F0u, 0x26000F0u, 0x26000F0u,   // room 1 (4-7)
    0x3800F00u, 0x3800F00u, 0x3800F00u, 0x3800F00u,   // room 2 (8-11)
    0x200F000u, 0x200F000u, 0x200F000u, 0x200F000u,   // room 3 (12-15)
    0x20F0000u, 0x20F0000u, 0x20F0000u, 0x20F0000u,   // room 4 (16-19)
    0x210000Fu,                                       // 20 -> room 5
    0x26000F0u, 0x26000F0u,                           // 21,22 -> room 1
    0x3800F00u, 0x3800F00u,                           // 23,24 -> room 2
    0x3FFFFFFu                                        // 25 -> global
};

union Frag16 { v16h v; v8h h[2]; };

// Row-major f16 matrix (LDS) -> WMMA 16x16x32 fragment.
__device__ __forceinline__ v16h frag_ld(const _Float16* base, int r0, int c0,
                                        int ld, int lane) {
    int r = r0 + (lane & 15);
    int c = c0 + ((lane >> 4) << 3);
    const _Float16* p = base + r * ld + c;
    Frag16 f;
    f.h[0] = *(const v8h*)(p);
    f.h[1] = *(const v8h*)(p + 16);
    return f.v;
}

// A-fragment of x straight from global fp32 (rows >= 26 zero-padded).
// Executed once per batch; results stay resident in VGPRs.
__device__ __forceinline__ v16h frag_ld_x(const float* xb, int r0, int c0,
                                          int lane) {
    int r = r0 + (lane & 15);
    int c = c0 + ((lane >> 4) << 3);
    float4 a = {0.f, 0.f, 0.f, 0.f}, b = a, cc = a, d = a;
    if (r < 26) {
        const float* p = xb + r * 256 + c;
        a  = *(const float4*)(p);
        b  = *(const float4*)(p + 4);
        cc = *(const float4*)(p + 16);
        d  = *(const float4*)(p + 20);
    }
    Frag16 f;
    f.h[0] = (v8h){(_Float16)a.x,  (_Float16)a.y,  (_Float16)a.z,  (_Float16)a.w,
                   (_Float16)b.x,  (_Float16)b.y,  (_Float16)b.z,  (_Float16)b.w};
    f.h[1] = (v8h){(_Float16)cc.x, (_Float16)cc.y, (_Float16)cc.z, (_Float16)cc.w,
                   (_Float16)d.x,  (_Float16)d.y,  (_Float16)d.z,  (_Float16)d.w};
    return f.v;
}

// Issue async copy of one 8 KB weight tile into LDS (2 x b128 per thread).
__device__ __forceinline__ void stage_issue(const _Float16* wf16, int tile,
                                            _Float16* dstlds, int tid) {
    unsigned lds0 = (unsigned)(uintptr_t)dstlds + (unsigned)tid * 16u;
    unsigned long long g0 =
        (unsigned long long)(uintptr_t)(wf16 + (size_t)tile * TILE_HALF) +
        (unsigned long long)tid * 16ull;
    asm volatile("global_load_async_to_lds_b128 %0, %1, off"
                 :: "v"(lds0), "v"(g0) : "memory");
    asm volatile("global_load_async_to_lds_b128 %0, %1, off offset:4096"
                 :: "v"(lds0), "v"(g0) : "memory");
}

// ---------- prologue: convert + tile-reorder weights to f16 in workspace ----------
// Tile t < 48: qkv tile, h=t/6, sel=(t%6)/2, nt=t%2, rows [sel*256+h*32+nt*16, +16)
// Tile t >= 48: out_w rows [(t-48)*16, +16).  Each tile = 4096 halves contiguous.
__global__ __launch_bounds__(256)
void convert_weights(const float* __restrict__ qkv_w,
                     const float* __restrict__ out_w,
                     _Float16* __restrict__ wf16) {
    int g = blockIdx.x * 256 + threadIdx.x;     // one float4 per thread
    int idx = g * 4;
    if (idx >= NTILES * TILE_HALF) return;
    int tile = idx >> 12;          // /4096
    int within = idx & 4095;
    const float* src;
    if (tile < 48) {
        int h = tile / 6, r = tile % 6, sel = r >> 1, nt = r & 1;
        int wrow = sel * 256 + h * 32 + nt * 16;
        src = qkv_w + (size_t)wrow * 256;
    } else {
        src = out_w + (size_t)(tile - 48) * 16 * 256;
    }
    float4 f = *(const float4*)(src + within);
    *(v4h*)(wf16 + idx) = (v4h){(_Float16)f.x, (_Float16)f.y,
                                (_Float16)f.z, (_Float16)f.w};
}

// ---------- main fused kernel ----------
__global__ __launch_bounds__(THREADS)
void brain_attn_fused(const float* __restrict__ x,
                      const _Float16* __restrict__ wf16,
                      const float* __restrict__ qkv_b,
                      const float* __restrict__ out_b,
                      float* __restrict__ out) {
    // ---- LDS (240 KB total; <= 320 KB/workgroup on CDNA5) ----
    __shared__ _Float16 wlds[2][TILE_HALF];      // double-buffered tile (16 KB)
    __shared__ _Float16 qbuf[NB][32 * 32];       // q  [s][d]           (16 KB)
    __shared__ _Float16 kbuf[NB][32 * 32];       // k  [s][d]           (16 KB)
    __shared__ _Float16 vtbuf[NB][32 * 32];      // v^T [d][s]          (16 KB)
    __shared__ float    sbuf[NB][32 * 32];       // scores f32          (32 KB)
    __shared__ _Float16 pbuf[NB][32 * 32];       // probs f16           (16 KB)
    __shared__ _Float16 ctxbuf[NB][32 * 256];    // ctx f16            (128 KB)

    const int tid  = threadIdx.x;
    const int lane = tid & 31;
    const int wv   = tid >> 5;                   // wave id = local batch id
    const int b    = blockIdx.x * NB + wv;
    const float* xb = x + (size_t)b * 26 * 256;

    const int nlo = lane & 15;                   // D-frag: N = lane%16
    const int mhi = (lane >> 4) << 3;            // D-frag: M base (0 or 8)

    // Preload the batch's x A-fragments once; resident in 128 VGPRs.
    v16h xa[2][8];
#pragma unroll
    for (int mt = 0; mt < 2; ++mt)
#pragma unroll
        for (int ks = 0; ks < 8; ++ks)
            xa[mt][ks] = frag_ld_x(xb, mt * 16, ks * 32, lane);

    // Prime the async pipeline with tile 0.
    stage_issue(wf16, 0, wlds[0], tid);

    for (int t = 0; t < NTILES; ++t) {
        const int cur = t & 1;
        if (t + 1 < NTILES) {
            stage_issue(wf16, t + 1, wlds[cur ^ 1], tid);
            // 2 just-issued may remain in flight; previous tile's 2 must land.
            asm volatile("s_wait_asynccnt 0x2" ::: "memory");
        } else {
            asm volatile("s_wait_asynccnt 0x0" ::: "memory");
        }
        __syncthreads();
        const _Float16* wt = wlds[cur];

        if (t < 48) {
            // ================= QKV tile =================
            const int h = t / 6, r = t % 6, sel = r >> 1, nt = r & 1;
            const int wrow = sel * 256 + h * 32 + nt * 16;
            const float bias = qkv_b[wrow + nlo];

            v8f acc[2] = {{}, {}};
#pragma unroll
            for (int ks = 0; ks < 8; ++ks) {
                v16h bf = frag_ld(wt, 0, ks * 32, 256, lane);
                acc[0] = __builtin_amdgcn_wmma_f32_16x16x32_f16(
                    false, xa[0][ks], false, bf, (short)0, acc[0], false, false);
                acc[1] = __builtin_amdgcn_wmma_f32_16x16x32_f16(
                    false, xa[1][ks], false, bf, (short)0, acc[1], false, false);
            }
#pragma unroll
            for (int mt = 0; mt < 2; ++mt) {
                if (sel == 2) {
                    // v stored transposed [d][s]: one ds_store_b128
                    v8h pk;
#pragma unroll
                    for (int i = 0; i < 8; ++i)
                        pk[i] = (_Float16)(acc[mt][i] + bias);
                    *(v8h*)(&vtbuf[wv][(nt * 16 + nlo) * 32 + mt * 16 + mhi]) = pk;
                } else {
                    _Float16* dst = (sel == 0) ? qbuf[wv] : kbuf[wv];
#pragma unroll
                    for (int i = 0; i < 8; ++i)
                        dst[(mt * 16 + mhi + i) * 32 + nt * 16 + nlo] =
                            (_Float16)(acc[mt][i] + bias);
                }
            }

            if (r == 5) {
                // ============ attention for head h (wave-private) ============
#pragma unroll
                for (int mt = 0; mt < 2; ++mt)
#pragma unroll
                    for (int nt2 = 0; nt2 < 2; ++nt2) {
                        v16h a  = frag_ld(qbuf[wv], mt * 16, 0, 32, lane);
                        v16h bf = frag_ld(kbuf[wv], nt2 * 16, 0, 32, lane);
                        v8f s = {};
                        s = __builtin_amdgcn_wmma_f32_16x16x32_f16(
                            false, a, false, bf, (short)0, s, false, false);
#pragma unroll
                        for (int i = 0; i < 8; ++i)
                            sbuf[wv][(mt * 16 + mhi + i) * 32 + nt2 * 16 + nlo] =
                                s[i] * 0.17677669529663687f;   // 1/sqrt(32)
                    }
                {   // masked softmax: lane = query row
                    const int m = lane;
                    float p[26];
                    float rinv = 0.f;
                    if (m < 26) {
                        const unsigned msk = MASKROW[m];
                        float mx = -3.0e38f;
#pragma unroll
                        for (int n = 0; n < 26; ++n)
                            if ((msk >> n) & 1u)
                                mx = fmaxf(mx, sbuf[wv][m * 32 + n]);
                        float s = 0.f;
#pragma unroll
                        for (int n = 0; n < 26; ++n) {
                            float e = ((msk >> n) & 1u)
                                          ? __expf(sbuf[wv][m * 32 + n] - mx) : 0.f;
                            p[n] = e;
                            s += e;
                        }
                        rinv = 1.0f / s;
                    }
#pragma unroll
                    for (int n = 0; n < 32; ++n) {
                        float val = (m < 26 && n < 26) ? p[n] * rinv : 0.f;
                        pbuf[wv][m * 32 + n] = (_Float16)val;
                    }
                }
                // ctx = probs @ v
#pragma unroll
                for (int mt = 0; mt < 2; ++mt)
#pragma unroll
                    for (int dt = 0; dt < 2; ++dt) {
                        v16h a  = frag_ld(pbuf[wv], mt * 16, 0, 32, lane);
                        v16h bf = frag_ld(vtbuf[wv], dt * 16, 0, 32, lane);
                        v8f c = {};
                        c = __builtin_amdgcn_wmma_f32_16x16x32_f16(
                            false, a, false, bf, (short)0, c, false, false);
#pragma unroll
                        for (int i = 0; i < 8; ++i)
                            ctxbuf[wv][(mt * 16 + mhi + i) * 256 +
                                       h * 32 + dt * 16 + nlo] = (_Float16)c[i];
                    }
            }
        } else {
            // ================= output-projection tile =================
            const int ont  = t - 48;
            const int fcol = ont * 16 + nlo;
            const float bias = out_b[fcol];
            v8f acc[2] = {{}, {}};
#pragma unroll
            for (int ks = 0; ks < 8; ++ks) {
                v16h bf = frag_ld(wt, 0, ks * 32, 256, lane);
                v16h a0 = frag_ld(ctxbuf[wv], 0,  ks * 32, 256, lane);
                v16h a1 = frag_ld(ctxbuf[wv], 16, ks * 32, 256, lane);
                acc[0] = __builtin_amdgcn_wmma_f32_16x16x32_f16(
                    false, a0, false, bf, (short)0, acc[0], false, false);
                acc[1] = __builtin_amdgcn_wmma_f32_16x16x32_f16(
                    false, a1, false, bf, (short)0, acc[1], false, false);
            }
#pragma unroll
            for (int mt = 0; mt < 2; ++mt)
#pragma unroll
                for (int i = 0; i < 8; ++i) {
                    int m = mt * 16 + mhi + i;
                    if (m < 26)
                        out[(size_t)b * 26 * 256 + (size_t)m * 256 + fcol] =
                            acc[mt][i] + bias;
                }
        }
        __syncthreads();   // release wlds[cur] for the t+2 issue
    }
}

extern "C" void kernel_launch(void* const* d_in, const int* in_sizes, int n_in,
                              void* d_out, int out_size, void* d_ws, size_t ws_size,
                              hipStream_t stream) {
    const float* x     = (const float*)d_in[0];
    const float* qkv_w = (const float*)d_in[1];
    const float* qkv_b = (const float*)d_in[2];
    const float* out_w = (const float*)d_in[3];
    const float* out_b = (const float*)d_in[4];
    float* out = (float*)d_out;
    _Float16* wf16 = (_Float16*)d_ws;            // 64 tiles * 8 KB = 512 KB

    convert_weights<<<dim3(NTILES * TILE_HALF / (4 * 256)), dim3(256), 0, stream>>>(
        qkv_w, out_w, wf16);

    const int B = 16384;
    brain_attn_fused<<<dim3(B / NB), dim3(THREADS), 0, stream>>>(
        x, wf16, qkv_b, out_b, out);
}
